// RelInteractionDebug3_27032524161362
// MI455X (gfx1250) — compile-verified, matrix operand order checked by reading, not run on hardware
//
#include <hip/hip_runtime.h>
#include <hip/hip_bf16.h>

// ---------------------------------------------------------------------------
// Problem constants (match reference)
// ---------------------------------------------------------------------------
#define CCH   512
#define HIMG  38
#define WIMG  38
#define HWPX  1444          // 38*38
#define NOBJ  64
#define NREL  256
#define NIMG  2
#define PP    49            // 7*7

typedef __attribute__((ext_vector_type(16))) __bf16 v16bf;
typedef __attribute__((ext_vector_type(8)))  float  v8f;

__device__ __forceinline__ unsigned short f2bf(float f) {
    // round-to-nearest-even fp32 -> bf16 (bit-level)
    unsigned int u = __float_as_uint(f);
    unsigned int r = u + 0x7FFFu + ((u >> 16) & 1u);
    return (unsigned short)(r >> 16);
}
__device__ __forceinline__ unsigned int pack2(float a, float b) {
    return (unsigned int)f2bf(a) | ((unsigned int)f2bf(b) << 16);
}

// ---------------------------------------------------------------------------
// Batched WMMA GEMM:  C[b] = act( A[b] (MxK) * B[b] (KxN) + bias )
//   !ATRANS : A element = A[m*lda + k]      (row-major)
//    ATRANS : A element = A[k*lda + m]      (K-major, e.g. W^T operands)
//   B element = B[k*ldb + n]
//   C row-major [M,N].
//   BIAS_MODE: 0 none, 1 per-N, 2 per-M.  BSCALE: B <- B*kinv[k]+kbeta[k].
//
// Block tile 64x64, BK=32, 256 threads = 8 wave32 (4 m-waves x 2 n-waves),
// each wave: one 16x32 slab -> two v_wmma_f32_16x16x32_bf16 per K-step.
// LDS: double-buffered; A row-per-m (32 bf16), B TRANSPOSED row-per-n,
// pitch 40 bf16 (80B) -> every fragment is two 16B-aligned ds_load_b128.
// Software pipeline: issue tile k+1 global_load_b128 BEFORE the WMMAs of
// tile k (latency hidden under matrix math); one barrier per iteration.
// Assumes M % 64 == 0 and K % 32 == 0 (true for every GEMM in this net);
// N tails (1444/49/51) take a guarded scalar path.
// ---------------------------------------------------------------------------
#define LDSP 40   // LDS pitch in bf16 elements (80 bytes)

template<int BIAS_MODE, bool RELU, bool BSCALE, bool ATRANS>
__global__ __launch_bounds__(256)
void wmma_gemm(const float* __restrict__ A, long long lda, long long sAb,
               const float* __restrict__ B, long long ldb, long long sBb,
               const float* __restrict__ bias,
               const float* __restrict__ kinv, const float* __restrict__ kbeta,
               float* __restrict__ Cmat, long long sCb,
               int M, int N, int K)
{
    __shared__ unsigned short As [2][64 * LDSP];   // [m][k]  32 bf16 per row
    __shared__ unsigned short BsT[2][64 * LDSP];   // [n][k]  32 bf16 per row

    const int tid  = threadIdx.x;
    const int wave = tid >> 5;
    const int lane = tid & 31;
    const int wm   = wave & 3;     // m sub-tile (16 rows)
    const int wn   = wave >> 2;    // 32-column group
    const int bm   = blockIdx.y * 64;
    const int bn   = blockIdx.x * 64;
    const bool nFull = (bn + 64 <= N);

    const float* Ab = A + (long long)blockIdx.z * sAb;
    const float* Bb = B + (long long)blockIdx.z * sBb;
    float*       Cb = Cmat + (long long)blockIdx.z * sCb;

    // staging coordinates (fixed per thread)
    const int a_row = tid >> 2;             // !ATRANS: 0..63
    const int a_kc  = (tid & 3) * 8;        // !ATRANS: 0,8,16,24
    const int a_kk  = tid >> 3;             //  ATRANS: 0..31
    const int a_mc  = (tid & 7) * 8;        //  ATRANS: 0..56
    const int b_kr  = tid >> 3;             // 0..31
    const int b_nc  = (tid & 7) * 8;        // 0..56

    float av[8], bv[8];
    float bsc = 1.f, bsh = 0.f;

    // ---- load tile k0 (global -> regs) ----
    auto load_tile = [&](int k0) {
        if (!ATRANS) {
            const float* ap = Ab + (long long)(bm + a_row) * lda + (k0 + a_kc);
            if (k0 + 32 < K) __builtin_prefetch(ap + 32, 0, 1);
            const float4 f0 = *(const float4*)ap;
            const float4 f1 = *(const float4*)(ap + 4);
            av[0]=f0.x; av[1]=f0.y; av[2]=f0.z; av[3]=f0.w;
            av[4]=f1.x; av[5]=f1.y; av[6]=f1.z; av[7]=f1.w;
        } else {
            const float* ap = Ab + (long long)(k0 + a_kk) * lda + (bm + a_mc);
            if (k0 + 32 < K) __builtin_prefetch(ap + 32 * lda, 0, 1);
            const float4 f0 = *(const float4*)ap;
            const float4 f1 = *(const float4*)(ap + 4);
            av[0]=f0.x; av[1]=f0.y; av[2]=f0.z; av[3]=f0.w;
            av[4]=f1.x; av[5]=f1.y; av[6]=f1.z; av[7]=f1.w;
        }
        const int kg = k0 + b_kr;
        if (BSCALE) { bsc = kinv[kg]; bsh = kbeta[kg]; }
        const float* bp = Bb + (long long)kg * ldb + (bn + b_nc);
        if (nFull) {
            if (k0 + 32 < K) __builtin_prefetch(bp + 32 * ldb, 0, 1);
            const float4 f0 = *(const float4*)bp;
            const float4 f1 = *(const float4*)(bp + 4);
            bv[0]=f0.x; bv[1]=f0.y; bv[2]=f0.z; bv[3]=f0.w;
            bv[4]=f1.x; bv[5]=f1.y; bv[6]=f1.z; bv[7]=f1.w;
        } else {
#pragma unroll
            for (int i = 0; i < 8; ++i)
                bv[i] = (bn + b_nc + i < N) ? bp[i] : 0.f;
        }
    };

    // ---- convert + store regs -> LDS buffer ----
    auto store_tile = [&](int buf) {
        if (!ATRANS) {
            *(uint4*)&As[buf][a_row * LDSP + a_kc] =
                make_uint4(pack2(av[0], av[1]), pack2(av[2], av[3]),
                           pack2(av[4], av[5]), pack2(av[6], av[7]));
        } else {
#pragma unroll
            for (int i = 0; i < 8; ++i)           // transpose scatter
                As[buf][(a_mc + i) * LDSP + a_kk] = f2bf(av[i]);
        }
#pragma unroll
        for (int i = 0; i < 8; ++i) {
            float v = bv[i];
            if (BSCALE) v = v * bsc + bsh;
            BsT[buf][(b_nc + i) * LDSP + b_kr] = f2bf(v);  // transpose scatter
        }
    };

    v8f acc0 = {}; v8f acc1 = {};

    // prologue: stage tile 0
    load_tile(0);
    store_tile(0);
    __syncthreads();

    const int kh = (lane >> 4) * 8;
    const int a_off = (wm * 16 + (lane & 15)) * LDSP + kh;
    const int b_off = (wn * 32 + (lane & 15)) * LDSP + kh;

    for (int k0 = 0; k0 < K; k0 += 32) {
        const int cur = (k0 >> 5) & 1;
        const bool more = (k0 + 32 < K);

        if (more) load_tile(k0 + 32);       // global loads in flight during WMMA

        // ---- fragments: two 16B ds_load_b128 each (ISA 7.12.2 bf16 layout) ----
        union Frag { uint4 q[2]; v16bf v; };
        Frag af, bf0, bf1;
        {
            const unsigned short* ar = &As[cur][a_off];
            af.q[0] = *(const uint4*)ar;
            af.q[1] = *(const uint4*)(ar + 16);
        }
        {
            const unsigned short* br = &BsT[cur][b_off];
            bf0.q[0] = *(const uint4*)br;
            bf0.q[1] = *(const uint4*)(br + 16);
            br += 16 * LDSP;
            bf1.q[0] = *(const uint4*)br;
            bf1.q[1] = *(const uint4*)(br + 16);
        }
        acc0 = __builtin_amdgcn_wmma_f32_16x16x32_bf16(false, af.v, false, bf0.v,
                                                       (short)0, acc0, false, false);
        acc1 = __builtin_amdgcn_wmma_f32_16x16x32_bf16(false, af.v, false, bf1.v,
                                                       (short)0, acc1, false, false);

        if (more) store_tile(cur ^ 1);      // fill the other buffer
        __syncthreads();                    // next-buf visible; cur safe to reuse
    }

    // ---- epilogue: C/D layout (lanes 0-15: M=vgpr, N=lane; 16-31: M=vgpr+8) ----
    {
        const int nlo  = lane & 15;
        const int mofs = (lane >> 4) * 8;
#pragma unroll
        for (int i = 0; i < 8; ++i) {
            const int m = bm + wm * 16 + mofs + i;     // always < M (M%64==0)
            float bmb = 0.f;
            if (BIAS_MODE == 2) bmb = bias[m];
            {
                const int n = bn + wn * 32 + nlo;
                if (n < N) {
                    float v = acc0[i] + bmb;
                    if (BIAS_MODE == 1) v += bias[n];
                    if (RELU) v = fmaxf(v, 0.f);
                    Cb[(long long)m * N + n] = v;
                }
            }
            {
                const int n = bn + wn * 32 + 16 + nlo;
                if (n < N) {
                    float v = acc1[i] + bmb;
                    if (BIAS_MODE == 1) v += bias[n];
                    if (RELU) v = fmaxf(v, 0.f);
                    Cb[(long long)m * N + n] = v;
                }
            }
        }
    }
}

// ---------------------------------------------------------------------------
// Small helper kernels
// ---------------------------------------------------------------------------
__global__ void zero_kernel(float* p, int n) {
    int i = blockIdx.x * blockDim.x + threadIdx.x;
    if (i < n) p[i] = 0.f;
}

// Depthwise 3x3 SAME conv with per-object generated filters + BN statistics
// grid: (ceil(HW/256), C, NOBJ)
__global__ __launch_bounds__(256)
void dwconv_stats(const float* __restrict__ rf, const float* __restrict__ dwf,
                  const int* __restrict__ im_inds,
                  float* __restrict__ dw, float* __restrict__ sum, float* __restrict__ sumsq)
{
    const int hw = blockIdx.x * 256 + threadIdx.x;
    const int c  = blockIdx.y;
    const int n  = blockIdx.z;
    float v = 0.f;
    if (hw < HWPX) {
        const int h = hw / WIMG, w = hw % WIMG;
        const int img = im_inds[n];
        const float* src = rf + ((long long)img * CCH + c) * HWPX;
        const float* kr  = dwf + (long long)n * (CCH * 9) + c * 9;
#pragma unroll
        for (int ky = 0; ky < 3; ++ky) {
            const int y = h + ky - 1;
            if (y < 0 || y >= HIMG) continue;
#pragma unroll
            for (int kx = 0; kx < 3; ++kx) {
                const int x = w + kx - 1;
                if (x < 0 || x >= WIMG) continue;
                v += src[y * WIMG + x] * kr[ky * 3 + kx];
            }
        }
        dw[((long long)n * CCH + c) * HWPX + hw] = v;
    }
    __shared__ float s1[256], s2[256];
    const float vv = (hw < HWPX) ? v : 0.f;
    s1[threadIdx.x] = vv; s2[threadIdx.x] = vv * vv;
    __syncthreads();
    for (int s = 128; s > 0; s >>= 1) {
        if (threadIdx.x < s) {
            s1[threadIdx.x] += s1[threadIdx.x + s];
            s2[threadIdx.x] += s2[threadIdx.x + s];
        }
        __syncthreads();
    }
    if (threadIdx.x == 0) { atomicAdd(&sum[c], s1[0]); atomicAdd(&sumsq[c], s2[0]); }
}

// per-(n,d) partial BN stats; grid: (C, NOBJ)
__global__ __launch_bounds__(256)
void chan_stats(const float* __restrict__ x, float* sum, float* sumsq)
{
    const int d = blockIdx.x, n = blockIdx.y;
    const float* p = x + ((long long)n * CCH + d) * HWPX;
    float a = 0.f, b = 0.f;
    for (int i = threadIdx.x; i < HWPX; i += 256) { float v = p[i]; a += v; b += v * v; }
    __shared__ float s1[256], s2[256];
    s1[threadIdx.x] = a; s2[threadIdx.x] = b;
    __syncthreads();
    for (int s = 128; s > 0; s >>= 1) {
        if (threadIdx.x < s) {
            s1[threadIdx.x] += s1[threadIdx.x + s];
            s2[threadIdx.x] += s2[threadIdx.x + s];
        }
        __syncthreads();
    }
    if (threadIdx.x == 0) { atomicAdd(&sum[d], s1[0]); atomicAdd(&sumsq[d], s2[0]); }
}

__global__ void finalize_stats(const float* sum, const float* sumsq,
                               const float* scale, const float* biasv,
                               float* inv, float* beta, float cnt)
{
    const int c = blockIdx.x * blockDim.x + threadIdx.x;
    if (c >= CCH) return;
    const float m   = sum[c] / cnt;
    const float var = sumsq[c] / cnt - m * m;
    const float iv  = scale[c] * rsqrtf(var + 1e-5f);
    inv[c]  = iv;
    beta[c] = biasv[c] - m * iv;
}

// ROI-align 7x7 with fused BN+relu of pw applied to each bilinear neighbor.
// grid: (ceil(C*49/256), 2, NREL)
__global__ __launch_bounds__(256)
void roi_kernel(const float* __restrict__ pw, const float* __restrict__ boxes,
                const int* __restrict__ rel_inds,
                const float* __restrict__ inv, const float* __restrict__ beta,
                float* __restrict__ SO)
{
    const int idx = blockIdx.x * 256 + threadIdx.x;     // c*49 + p
    if (idx >= CCH * PP) return;
    const int side = blockIdx.y;
    const int r    = blockIdx.z;
    const int c  = idx / PP;
    const int p  = idx % PP;
    const int py = p / 7, px = p % 7;
    const int obj = rel_inds[r * 2 + side];
    const float x1 = boxes[obj * 4 + 0] * (1.f / 16.f);
    const float y1 = boxes[obj * 4 + 1] * (1.f / 16.f);
    const float x2 = boxes[obj * 4 + 2] * (1.f / 16.f);
    const float y2 = boxes[obj * 4 + 3] * (1.f / 16.f);
    const float ys = y1 + (y2 - y1) * ((py + 0.5f) / 7.f);
    const float xs = x1 + (x2 - x1) * ((px + 0.5f) / 7.f);
    const float y0f = floorf(ys), x0f = floorf(xs);
    const float wy = ys - y0f, wx = xs - x0f;
    const int y0i = min(max((int)y0f, 0), HIMG - 1);
    const int y1i = min(y0i + 1, HIMG - 1);
    const int x0i = min(max((int)x0f, 0), WIMG - 1);
    const int x1i = min(x0i + 1, WIMG - 1);
    const float* fm = pw + ((long long)obj * CCH + c) * HWPX;
    const float iv = inv[c], bt = beta[c];
    const float v00 = fmaxf(fm[y0i * WIMG + x0i] * iv + bt, 0.f);
    const float v01 = fmaxf(fm[y0i * WIMG + x1i] * iv + bt, 0.f);
    const float v10 = fmaxf(fm[y1i * WIMG + x0i] * iv + bt, 0.f);
    const float v11 = fmaxf(fm[y1i * WIMG + x1i] * iv + bt, 0.f);
    const float val = v00 * (1.f - wy) * (1.f - wx) + v01 * (1.f - wy) * wx
                    + v10 * wy * (1.f - wx)         + v11 * wy * wx;
    SO[((long long)r * (2 * CCH) + side * CCH + c) * PP + p] = val;
}

// cat[r, j] = subobj[ rel_inds[r, j>=512], j ]
__global__ void gather_cat(const float* __restrict__ subobj, const int* __restrict__ rel_inds,
                           float* __restrict__ cat)
{
    const int i = blockIdx.x * 256 + threadIdx.x;
    if (i >= NREL * 1024) return;
    const int r = i >> 10, j = i & 1023;
    const int obj = rel_inds[r * 2 + (j >> 9)];
    cat[i] = subobj[obj * 1024 + j];
}

__global__ void triple_kernel(const float* __restrict__ lo, const float* __restrict__ lr,
                              float* __restrict__ t, int n)
{
    const int i = blockIdx.x * 256 + threadIdx.x;
    if (i >= n) return;
    const float a = lo[i], b = lr[i];
    const float d = a - b;
    t[i] = fmaxf(a + b, 0.f) - d * d;
}

// ---------------------------------------------------------------------------
static inline dim3 gemm_grid(int M, int N, int batch) {
    return dim3((unsigned)((N + 63) / 64), (unsigned)((M + 63) / 64), (unsigned)batch);
}

extern "C" void kernel_launch(void* const* d_in, const int* in_sizes, int n_in,
                              void* d_out, int out_size, void* d_ws, size_t ws_size,
                              hipStream_t stream)
{
    (void)in_sizes; (void)n_in; (void)out_size; (void)ws_size;
    const float* fmaps       = (const float*)d_in[0];
    const float* boxes       = (const float*)d_in[1];
    const float* feats       = (const float*)d_in[2];
    const float* reduce_w    = (const float*)d_in[3];
    const float* reduce_b    = (const float*)d_in[4];
    const float* dw_gen_w    = (const float*)d_in[5];
    const float* dw_gen_b    = (const float*)d_in[6];
    const float* pw_gen_w    = (const float*)d_in[7];
    const float* pw_gen_b    = (const float*)d_in[8];
    const float* dw_bn_scale = (const float*)d_in[9];
    const float* dw_bn_bias  = (const float*)d_in[10];
    const float* pw_bn_scale = (const float*)d_in[11];
    const float* pw_bn_bias  = (const float*)d_in[12];
    const float* recover_w   = (const float*)d_in[13];
    const float* recover_b   = (const float*)d_in[14];
    const float* fc1_w       = (const float*)d_in[15];
    const float* fc1_b       = (const float*)d_in[16];
    const float* fc2_w       = (const float*)d_in[17];
    const float* fc2_b       = (const float*)d_in[18];
    const float* post_obj_w  = (const float*)d_in[19];
    const float* post_obj_b  = (const float*)d_in[20];
    const float* mapping_w   = (const float*)d_in[21];
    const float* mapping_b   = (const float*)d_in[22];
    const float* rel_red_w   = (const float*)d_in[23];
    const float* rel_red_b   = (const float*)d_in[24];
    const float* comp_w      = (const float*)d_in[25];
    const float* comp_b      = (const float*)d_in[26];
    const int*   im_inds     = (const int*)d_in[27];
    const int*   rel_inds    = (const int*)d_in[28];
    float* out = (float*)d_out;

    // workspace layout (fp32 elements)
    float* ws = (float*)d_ws;
    size_t off = 0;
    float* dwf_ws  = ws + off; off += (size_t)NOBJ * 4608;
    float* pwf_ws  = ws + off; off += (size_t)NOBJ * CCH * CCH;
    float* rf_ws   = ws + off; off += (size_t)NIMG * CCH * HWPX;
    float* dw_ws   = ws + off; off += (size_t)NOBJ * CCH * HWPX;
    float* pw_ws   = ws + off; off += (size_t)NOBJ * CCH * HWPX;
    float* stats   = ws + off; off += 8 * CCH;
    float* dw_sum = stats;            float* dw_sumsq = stats + CCH;
    float* dw_inv = stats + 2 * CCH;  float* dw_beta  = stats + 3 * CCH;
    float* pw_sum = stats + 4 * CCH;  float* pw_sumsq = stats + 5 * CCH;
    float* pw_inv = stats + 6 * CCH;  float* pw_beta  = stats + 7 * CCH;
    float* so_ws   = ws + off; off += (size_t)NREL * 2 * CCH * PP;
    float* rec_ws  = ws + off; off += (size_t)NREL * CCH * PP;
    float* h1_ws   = ws + off; off += (size_t)NREL * 4096;
    float* ri_ws   = ws + off; off += (size_t)NREL * 4096;
    float* sub_ws  = ws + off; off += (size_t)NOBJ * 1024;
    float* cat_ws  = ws + off; off += (size_t)NREL * 1024;
    float* lo_ws   = ws + off; off += (size_t)NREL * 1536;
    float* lr_ws   = ws + off; off += (size_t)NREL * 1536;
    float* tr_ws   = ws + off; off += (size_t)NREL * 1536;

    const long long sPlane = (long long)CCH * HWPX;

    // 1) dw filter generation: [64,512] @ [512,4608] + b
    wmma_gemm<1, false, false, false><<<gemm_grid(NOBJ, 4608, 1), 256, 0, stream>>>(
        feats, 512, 0,  dw_gen_w, 4608, 0,
        dw_gen_b, nullptr, nullptr, dwf_ws, 0, NOBJ, 4608, 512);

    // 2) pw filter generation: [64,512] @ [512,262144] + b  (537 MB weights, BW-bound)
    wmma_gemm<1, false, false, false><<<gemm_grid(NOBJ, CCH * CCH, 1), 256, 0, stream>>>(
        feats, 512, 0,  pw_gen_w, (long long)CCH * CCH, 0,
        pw_gen_b, nullptr, nullptr, pwf_ws, 0, NOBJ, CCH * CCH, 512);

    // 3) rf = relu(reduce_w^T @ fmap[b] + reduce_b), per image (A transposed)
    wmma_gemm<2, true, false, true><<<gemm_grid(CCH, HWPX, NIMG), 256, 0, stream>>>(
        reduce_w, CCH, 0,  fmaps, HWPX, sPlane,
        reduce_b, nullptr, nullptr, rf_ws, sPlane, CCH, HWPX, CCH);

    // 4) zero BN accumulators
    zero_kernel<<<(8 * CCH + 255) / 256, 256, 0, stream>>>(stats, 8 * CCH);

    // 5) depthwise 3x3 + BN stats, then finalize
    dwconv_stats<<<dim3((HWPX + 255) / 256, CCH, NOBJ), 256, 0, stream>>>(
        rf_ws, dwf_ws, im_inds, dw_ws, dw_sum, dw_sumsq);
    finalize_stats<<<2, 256, 0, stream>>>(dw_sum, dw_sumsq, dw_bn_scale, dw_bn_bias,
                                          dw_inv, dw_beta, (float)(NOBJ * HWPX));

    // 6) pw[n] = pw_f[n] @ bn(dw[n]) — BN fused into B-tile load (dw read once)
    wmma_gemm<0, false, true, false><<<gemm_grid(CCH, HWPX, NOBJ), 256, 0, stream>>>(
        pwf_ws, CCH, (long long)CCH * CCH,  dw_ws, HWPX, sPlane,
        nullptr, dw_inv, dw_beta, pw_ws, sPlane, CCH, HWPX, CCH);

    // 7) pw BN stats + finalize (BN+relu applied inside ROI sampler)
    chan_stats<<<dim3(CCH, NOBJ), 256, 0, stream>>>(pw_ws, pw_sum, pw_sumsq);
    finalize_stats<<<2, 256, 0, stream>>>(pw_sum, pw_sumsq, pw_bn_scale, pw_bn_bias,
                                          pw_inv, pw_beta, (float)(NOBJ * HWPX));

    // 8) ROI align 7x7 for subj/obj with fused BN+relu
    roi_kernel<<<dim3((CCH * PP + 255) / 256, 2, NREL), 256, 0, stream>>>(
        pw_ws, boxes, rel_inds, pw_inv, pw_beta, so_ws);

    // 9) recover: per-rel recover_w^T (512x1024) @ SO[r] (1024x49) + b (per-M)
    wmma_gemm<2, false, false, true><<<gemm_grid(CCH, PP, NREL), 256, 0, stream>>>(
        recover_w, CCH, 0,  so_ws, PP, (long long)2 * CCH * PP,
        recover_b, nullptr, nullptr, rec_ws, (long long)CCH * PP, CCH, PP, 2 * CCH);

    // 10) fc1: relu([256,25088] @ [25088,4096] + b)
    wmma_gemm<1, true, false, false><<<gemm_grid(NREL, 4096, 1), 256, 0, stream>>>(
        rec_ws, (long long)CCH * PP, 0,  fc1_w, 4096, 0,
        fc1_b, nullptr, nullptr, h1_ws, 0, NREL, 4096, CCH * PP);

    // 11) fc2
    wmma_gemm<1, false, false, false><<<gemm_grid(NREL, 4096, 1), 256, 0, stream>>>(
        h1_ws, 4096, 0,  fc2_w, 4096, 0,
        fc2_b, nullptr, nullptr, ri_ws, 0, NREL, 4096, 4096);

    // 12) subobj = feats @ post_obj_w + b
    wmma_gemm<1, false, false, false><<<gemm_grid(NOBJ, 1024, 1), 256, 0, stream>>>(
        feats, 512, 0,  post_obj_w, 1024, 0,
        post_obj_b, nullptr, nullptr, sub_ws, 0, NOBJ, 1024, 512);

    // 13) gather + concat sub/obj halves per relation
    gather_cat<<<(NREL * 1024 + 255) / 256, 256, 0, stream>>>(sub_ws, rel_inds, cat_ws);

    // 14) last_obj = cat @ mapping_w + b
    wmma_gemm<1, false, false, false><<<gemm_grid(NREL, 1536, 1), 256, 0, stream>>>(
        cat_ws, 1024, 0,  mapping_w, 1536, 0,
        mapping_b, nullptr, nullptr, lo_ws, 0, NREL, 1536, 1024);

    // 15) last_rel = rel_input @ rel_red_w + b
    wmma_gemm<1, false, false, false><<<gemm_grid(NREL, 1536, 1), 256, 0, stream>>>(
        ri_ws, 4096, 0,  rel_red_w, 1536, 0,
        rel_red_b, nullptr, nullptr, lr_ws, 0, NREL, 1536, 4096);

    // 16) triple = relu(lo+lr) - (lo-lr)^2
    triple_kernel<<<(NREL * 1536 + 255) / 256, 256, 0, stream>>>(
        lo_ws, lr_ws, tr_ws, NREL * 1536);

    // 17) out = triple @ comp_w + comp_b   (N=51 tail guarded)
    wmma_gemm<1, false, false, false><<<gemm_grid(NREL, 51, 1), 256, 0, stream>>>(
        tr_ws, 1536, 0,  comp_w, 51, 0,
        comp_b, nullptr, nullptr, out, 0, NREL, 51, 1536);
}
